// GPTModel_21792664060261
// MI455X (gfx1250) — matrix operand order repaired and checked by reading, not compiled, not run on hardware
//
#include <hip/hip_runtime.h>
#include <hip/hip_bf16.h>
#include <math.h>

// ---------------- problem constants ----------------
#define Vv    50257
#define Ee    256
#define Hh    8
#define DHd   32
#define Bb    2
#define Tt    2048
#define SCALE_F 0.17677669529663689f   // 1/sqrt(32)

typedef __bf16 bf16_t;
typedef __attribute__((ext_vector_type(8)))  __bf16 v8bf;
typedef __attribute__((ext_vector_type(16))) __bf16 v16bf;
typedef __attribute__((ext_vector_type(8)))  float  v8f;

#define WMMA_BF16(A, B, C) \
  __builtin_amdgcn_wmma_f32_16x16x32_bf16(false, (A), false, (B), (short)0, (C), false, false)

// A-operand (16x32 bf16): lane m holds K 0..15 (low half) / 16..31 (high half), contiguous
__device__ __forceinline__ v16bf load_matA(const bf16_t* row, int kb, int half) {
  const bf16_t* p = row + kb + half * 16;
  v8bf lo = *(const v8bf*)(p);
  v8bf hi = *(const v8bf*)(p + 8);
  return __builtin_shufflevector(lo, hi, 0,1,2,3,4,5,6,7,8,9,10,11,12,13,14,15);
}
// B-operand (32x16 bf16): lane n holds K 0..15 (low half) / 16..31 (high half), contiguous
__device__ __forceinline__ v16bf load_matB(const bf16_t* row, int kb, int half) {
  return *(const v16bf*)(row + kb + half * 16);
}
__device__ __forceinline__ v8bf shfl16_v8bf(v8bf v) {
  union { v8bf b; unsigned u[4]; } x; x.b = v;
  #pragma unroll
  for (int i = 0; i < 4; i++) x.u[i] = __shfl_xor(x.u[i], 16, 32);
  return x.b;
}

// ---------------- K1: x = tok_emb[idx] + pos_emb, to bf16 ----------------
__global__ void __launch_bounds__(256) k_embed(const int* __restrict__ idx,
                                               const float* __restrict__ tok,
                                               const float* __restrict__ pos,
                                               bf16_t* __restrict__ xh) {
  int bt = blockIdx.x;              // 0..B*T-1
  int e  = threadIdx.x;             // 0..255
  int t  = bt % Tt;
  int id = idx[bt];
  float v = tok[(size_t)id * Ee + e] + pos[(size_t)t * Ee + e];
  xh[(size_t)bt * Ee + e] = (bf16_t)v;
}

// ---------------- K2: weights -> bf16 (Wq/Wk/Wv transposed to [H][DH][E]) ----------------
__global__ void __launch_bounds__(256) k_wconv(const float* __restrict__ Wq,
                                               const float* __restrict__ Wk,
                                               const float* __restrict__ Wv,
                                               const float* __restrict__ Wlm,
                                               bf16_t* __restrict__ wqT, bf16_t* __restrict__ wkT,
                                               bf16_t* __restrict__ wvT, bf16_t* __restrict__ wlmb) {
  long i = (long)blockIdx.x * 256 + threadIdx.x;
  const long NW = 3L * Hh * DHd * Ee;
  const long NL = (long)Vv * Ee;
  if (i < NW) {
    int mat = (int)(i / (Hh * DHd * Ee));
    int r   = (int)(i % (Hh * DHd * Ee));
    int h = r / (DHd * Ee); int r2 = r % (DHd * Ee);
    int d = r2 / Ee;        int e  = r2 % Ee;
    const float* W = (mat == 0) ? Wq : (mat == 1) ? Wk : Wv;
    bf16_t* dst    = (mat == 0) ? wqT : (mat == 1) ? wkT : wvT;
    dst[((long)h * DHd + d) * Ee + e] = (bf16_t)W[((long)h * Ee + e) * DHd + d];
  } else if (i < NW + NL) {
    long j = i - NW;
    wlmb[j] = (bf16_t)Wlm[j];
  }
}

// ---------------- K3: q/k/v = x @ W  (bf16 WMMA, f32 accum) ----------------
__global__ void __launch_bounds__(256) k_qkv(const bf16_t* __restrict__ xh,
                                             const bf16_t* __restrict__ wqT,
                                             const bf16_t* __restrict__ wkT,
                                             const bf16_t* __restrict__ wvT,
                                             bf16_t* __restrict__ qh, bf16_t* __restrict__ kh,
                                             bf16_t* __restrict__ vt) {
  int gw   = blockIdx.x * 8 + (threadIdx.x >> 5);
  int lane = threadIdx.x & 31;
  int n = lane & 15, half = lane >> 4;
  const int TT = Tt / 16;
  int tt = gw % TT; int r1 = gw / TT;
  int h  = r1 % Hh; int r2 = r1 / Hh;
  int b  = r2 % Bb; int mat = r2 / Bb;     // 0=q 1=k 2=v
  const bf16_t* wT   = (mat == 0) ? wqT : (mat == 1) ? wkT : wvT;
  const bf16_t* aRow = xh + (size_t)(b * Tt + tt * 16 + n) * Ee;
  size_t bh = (size_t)(b * Hh + h);
  #pragma unroll
  for (int nt = 0; nt < 2; nt++) {
    const bf16_t* bRow = wT + ((size_t)h * DHd + nt * 16 + n) * Ee;
    v8f acc = {};
    #pragma unroll
    for (int ks = 0; ks < 8; ks++) {
      v16bf A  = load_matA(aRow, ks * 32, half);
      v16bf Bm = load_matB(bRow, ks * 32, half);
      acc = WMMA_BF16(A, Bm, acc);
    }
    int d = nt * 16 + n;
    #pragma unroll
    for (int r = 0; r < 8; r++) {
      int t = tt * 16 + r + 8 * half;
      float val = acc[r];
      if (mat == 0)      qh[(bh * Tt + t) * DHd + d] = (bf16_t)val;
      else if (mat == 1) kh[(bh * Tt + t) * DHd + d] = (bf16_t)val;
      else               vt[(bh * DHd + d) * Tt + t] = (bf16_t)val;  // v transposed
    }
  }
}

// ---------------- K4: column softmax stats (softmax over QUERY axis, faithful mask bug) --------
__global__ void __launch_bounds__(256) k_colstats(const bf16_t* __restrict__ qh,
                                                  const bf16_t* __restrict__ kh,
                                                  float* __restrict__ mcol,
                                                  float* __restrict__ rzcol) {
  int gw   = blockIdx.x * 8 + (threadIdx.x >> 5);
  int lane = threadIdx.x & 31;
  int n = lane & 15, half = lane >> 4;
  const int KT = Tt / 16;
  int kt = gw % KT; int r1 = gw / KT;
  int h  = r1 % Hh; int b = r1 / Hh;
  size_t bh = (size_t)(b * Hh + h);
  int kcol = kt * 16 + n;
  const bf16_t* kRow = kh + (bh * Tt + kcol) * DHd;   // B operand: column k
  v16bf Bm = load_matB(kRow, 0, half);
  float mrun = -INFINITY, srun = 0.f;
  for (int qt = 0; qt < Tt / 16; qt++) {
    const bf16_t* aRow = qh + (bh * Tt + qt * 16 + n) * DHd;
    v16bf A = load_matA(aRow, 0, half);
    v8f c = {};
    c = WMMA_BF16(A, Bm, c);
    float vals[8]; float tmax = -INFINITY;
    #pragma unroll
    for (int r = 0; r < 8; r++) {
      int q = qt * 16 + r + 8 * half;
      float v = c[r] * SCALE_F;
      if (kcol > q)  v = 0.f;          // tril mask -> 0
      if (v == 0.f)  v = -INFINITY;    // masked_fill(scores==0, -inf)
      vals[r] = v; tmax = fmaxf(tmax, v);
    }
    tmax = fmaxf(tmax, __shfl_xor(tmax, 16, 32));
    float nm = fmaxf(mrun, tmax);
    if (nm != -INFINITY) {
      float p = 0.f;
      #pragma unroll
      for (int r = 0; r < 8; r++) p += __expf(vals[r] - nm);
      p += __shfl_xor(p, 16, 32);
      srun = srun * __expf(mrun - nm) + p;
      mrun = nm;
    }
  }
  if (half == 0) {
    mcol[bh * Tt + kcol]  = mrun;
    rzcol[bh * Tt + kcol] = (srun > 0.f) ? 1.f / srun : 0.f;
  }
}

// ---------------- K5: out^T = v^T @ attn^T  (recompute S^T via WMMA, attn never stored) -------
__global__ void __launch_bounds__(256) k_attnout(const bf16_t* __restrict__ qh,
                                                 const bf16_t* __restrict__ kh,
                                                 const bf16_t* __restrict__ vt,
                                                 const float* __restrict__ mcol,
                                                 const float* __restrict__ rzcol,
                                                 bf16_t* __restrict__ outh) {
  __shared__ float smM[Tt];
  __shared__ float smRZ[Tt];
  int gw   = blockIdx.x * 8 + (threadIdx.x >> 5);
  int lane = threadIdx.x & 31;
  int n = lane & 15, half = lane >> 4;
  const int QT = Tt / 16;
  int qt = gw % QT; int r1 = gw / QT;
  int h  = r1 % Hh; int b = r1 / Hh;
  size_t bh = (size_t)(b * Hh + h);
  // all 8 waves in this block share (b,h): stage column stats in LDS
  for (int i = threadIdx.x; i < Tt; i += 256) {
    smM[i]  = mcol[bh * Tt + i];
    smRZ[i] = rzcol[bh * Tt + i];
  }
  __syncthreads();
  int q = qt * 16 + n;
  const bf16_t* qRow = qh + (bh * Tt + q) * DHd;
  v16bf Bq = load_matB(qRow, 0, half);                 // fixed B = Q^T tile
  const bf16_t* vtBase = vt + bh * DHd * Tt;
  const bf16_t* vRow0  = vtBase + (size_t)(n)      * Tt;
  const bf16_t* vRow1  = vtBase + (size_t)(16 + n) * Tt;
  v8f acc0 = {}, acc1 = {};
  for (int kb = 0; kb < Tt; kb += 32) {
    const bf16_t* kRow0 = kh + (bh * Tt + kb + n)      * DHd;
    const bf16_t* kRow1 = kh + (bh * Tt + kb + 16 + n) * DHd;
    v16bf A0 = load_matA(kRow0, 0, half);
    v16bf A1 = load_matA(kRow1, 0, half);
    v8f st0 = {}, st1 = {};
    st0 = WMMA_BF16(A0, Bq, st0);                      // S^T[kb..kb+15][q-tile]
    st1 = WMMA_BF16(A1, Bq, st1);                      // S^T[kb+16..kb+31][q-tile]
    v8bf a0, a1;
    #pragma unroll
    for (int r = 0; r < 8; r++) {
      int k0 = kb + r + 8 * half;
      float v0 = st0[r] * SCALE_F;
      if (k0 > q)   v0 = 0.f;
      if (v0 == 0.f) v0 = -INFINITY;
      float av0 = (v0 == -INFINITY) ? 0.f : __expf(v0 - smM[k0]) * smRZ[k0];
      a0[r] = (bf16_t)av0;
      int k1 = k0 + 16;
      float v1 = st1[r] * SCALE_F;
      if (k1 > q)   v1 = 0.f;
      if (v1 == 0.f) v1 = -INFINITY;
      float av1 = (v1 == -INFINITY) ? 0.f : __expf(v1 - smM[k1]) * smRZ[k1];
      a1[r] = (bf16_t)av1;
    }
    // C-layout f32 -> B-operand bf16 layout: lane<->lane^16 swap of packed halves
    v8bf o0 = shfl16_v8bf(a0);
    v8bf o1 = shfl16_v8bf(a1);
    v8bf blo = half ? o1 : a0;   // K 0..7 (half 0) / 16..23 (half 1) of this 32-k slab
    v8bf bhi = half ? a1 : o0;   // K 8..15 (half 0) / 24..31 (half 1)
    v16bf Battn = __builtin_shufflevector(blo, bhi, 0,1,2,3,4,5,6,7,8,9,10,11,12,13,14,15);
    v16bf Av0 = load_matA(vRow0, kb, half);
    v16bf Av1 = load_matA(vRow1, kb, half);
    acc0 = WMMA_BF16(Av0, Battn, acc0);
    acc1 = WMMA_BF16(Av1, Battn, acc1);
  }
  size_t rowoff = (size_t)(b * Tt + q) * Ee + h * DHd;  // concat heads: E index = h*DH + d
  #pragma unroll
  for (int r = 0; r < 8; r++) {
    outh[rowoff + r + 8 * half]      = (bf16_t)acc0[r];
    outh[rowoff + 16 + r + 8 * half] = (bf16_t)acc1[r];
  }
}

// ---------------- K6: logits = out @ Wlm^T + blm  (the 105 GF / 824 MB stage) ----------------
#define TGT 4
__global__ void __launch_bounds__(256) k_lmhead(const bf16_t* __restrict__ outh,
                                                const bf16_t* __restrict__ wlmb,
                                                const float* __restrict__ blm,
                                                float* __restrict__ logits) {
  int gw   = blockIdx.x * 8 + (threadIdx.x >> 5);
  int lane = threadIdx.x & 31;
  int n = lane & 15, half = lane >> 4;
  const int NVT = (Vv + 15) / 16;          // 3142
  const int NTG = (Bb * Tt) / (16 * TGT);  // 64
  int vt_ = gw % NVT; int tg = gw / NVT;
  if (tg >= NTG) return;
  int v = vt_ * 16 + n;
  bool vok = (v < Vv);
  const bf16_t* bRow = wlmb + (size_t)(vok ? v : 0) * Ee;
  float bias = vok ? blm[v] : 0.f;
  v8f acc[TGT];
  const bf16_t* aRow[TGT];
  #pragma unroll
  for (int i = 0; i < TGT; i++) {
    #pragma unroll
    for (int r = 0; r < 8; r++) acc[i][r] = bias;
    aRow[i] = outh + (size_t)(tg * (16 * TGT) + i * 16 + n) * Ee;
  }
  #pragma unroll
  for (int ks = 0; ks < 8; ks++) {
    v16bf Bm;
    if (vok) Bm = load_matB(bRow, ks * 32, half);
    else     Bm = v16bf{};
    if (ks < 7) __builtin_prefetch(bRow + (ks + 1) * 32, 0, 1);
    #pragma unroll
    for (int i = 0; i < TGT; i++) {
      v16bf A = load_matA(aRow[i], ks * 32, half);
      acc[i] = WMMA_BF16(A, Bm, acc[i]);
    }
  }
  if (vok) {
    #pragma unroll
    for (int i = 0; i < TGT; i++)
      #pragma unroll
      for (int r = 0; r < 8; r++) {
        int t = tg * (16 * TGT) + i * 16 + r + 8 * half;
        logits[(size_t)t * Vv + v] = acc[i][r];
      }
  }
}

// ---------------- K7: loss = -mean log_softmax(logits)[target] ----------------
__global__ void k_loss_init(float* loss) { if (threadIdx.x == 0 && blockIdx.x == 0) *loss = 0.f; }

__global__ void __launch_bounds__(256) k_loss(const float* __restrict__ logits,
                                              const int* __restrict__ targets,
                                              float* __restrict__ loss) {
  __shared__ float smm[256];
  __shared__ float sms[256];
  int row = blockIdx.x;                 // 0..B*T-1
  const float* lp = logits + (size_t)row * Vv;
  float m = -INFINITY, s = 0.f;
  for (int v = threadIdx.x; v < Vv; v += 256) {
    float l  = lp[v];
    float nm = fmaxf(m, l);
    s = s * __expf(m - nm) + __expf(l - nm);
    m = nm;
  }
  smm[threadIdx.x] = m; sms[threadIdx.x] = s;
  __syncthreads();
  for (int stride = 128; stride > 0; stride >>= 1) {
    if ((int)threadIdx.x < stride) {
      float m2 = smm[threadIdx.x + stride], s2 = sms[threadIdx.x + stride];
      float m1 = smm[threadIdx.x],          s1 = sms[threadIdx.x];
      float nm = fmaxf(m1, m2);
      smm[threadIdx.x] = nm;
      sms[threadIdx.x] = s1 * __expf(m1 - nm) + s2 * __expf(m2 - nm);
    }
    __syncthreads();
  }
  if (threadIdx.x == 0) {
    float M = smm[0], S = sms[0];
    float lt = lp[targets[row]];
    float logp = lt - M - logf(S);
    atomicAdd(loss, -logp * (1.0f / (Bb * Tt)));
  }
}

// ---------------- launcher ----------------
extern "C" void kernel_launch(void* const* d_in, const int* in_sizes, int n_in,
                              void* d_out, int out_size, void* d_ws, size_t ws_size,
                              hipStream_t stream) {
  const int*   idx  = (const int*)d_in[0];
  const int*   tgt  = (const int*)d_in[1];
  const float* tok  = (const float*)d_in[2];
  const float* pos  = (const float*)d_in[3];
  const float* Wq   = (const float*)d_in[4];
  const float* Wk   = (const float*)d_in[5];
  const float* Wv   = (const float*)d_in[6];
  const float* Wlm  = (const float*)d_in[7];
  const float* blm  = (const float*)d_in[8];
  float* logits = (float*)d_out;
  float* loss   = logits + (size_t)Bb * Tt * Vv;

  // workspace carve (bf16 unless noted), all 256B aligned
  char* p = (char*)d_ws;
  auto take = [&](size_t elems, size_t esz) {
    void* r = (void*)p;
    size_t bytes = (elems * esz + 255) & ~(size_t)255;
    p += bytes;
    return r;
  };
  bf16_t* xh   = (bf16_t*)take((size_t)Bb * Tt * Ee, 2);
  bf16_t* wqT  = (bf16_t*)take((size_t)Hh * DHd * Ee, 2);
  bf16_t* wkT  = (bf16_t*)take((size_t)Hh * DHd * Ee, 2);
  bf16_t* wvT  = (bf16_t*)take((size_t)Hh * DHd * Ee, 2);
  bf16_t* wlmb = (bf16_t*)take((size_t)Vv * Ee, 2);
  bf16_t* qh   = (bf16_t*)take((size_t)Bb * Hh * Tt * DHd, 2);
  bf16_t* kh   = (bf16_t*)take((size_t)Bb * Hh * Tt * DHd, 2);
  bf16_t* vt   = (bf16_t*)take((size_t)Bb * Hh * DHd * Tt, 2);
  bf16_t* outh = (bf16_t*)take((size_t)Bb * Tt * Ee, 2);
  float*  mcol = (float*)take((size_t)Bb * Hh * Tt, 4);
  float*  rzcl = (float*)take((size_t)Bb * Hh * Tt, 4);

  k_embed<<<Bb * Tt, 256, 0, stream>>>(idx, tok, pos, xh);

  long nconv = 3L * Hh * DHd * Ee + (long)Vv * Ee;
  k_wconv<<<(unsigned)((nconv + 255) / 256), 256, 0, stream>>>(Wq, Wk, Wv, Wlm, wqT, wkT, wvT, wlmb);

  // 3 * B * H * (T/16) waves, 8 waves/block
  k_qkv<<<(3 * Bb * Hh * (Tt / 16)) / 8, 256, 0, stream>>>(xh, wqT, wkT, wvT, qh, kh, vt);

  // B * H * (T/16) waves
  k_colstats<<<(Bb * Hh * (Tt / 16)) / 8, 256, 0, stream>>>(qh, kh, mcol, rzcl);
  k_attnout <<<(Bb * Hh * (Tt / 16)) / 8, 256, 0, stream>>>(qh, kh, vt, mcol, rzcl, outh);

  // (B*T/(16*TGT)) * ceil(V/16) waves, 8 waves/block
  int nvt = (Vv + 15) / 16;
  int ntg = (Bb * Tt) / (16 * TGT);
  int lmwaves = nvt * ntg;
  k_lmhead<<<(lmwaves + 7) / 8, 256, 0, stream>>>(outh, wlmb, blm, logits);

  k_loss_init<<<1, 64, 0, stream>>>(loss);
  k_loss<<<Bb * Tt, 256, 0, stream>>>(logits, tgt, loss);
}
